// LSTMDQN_42889543417981
// MI455X (gfx1250) — matrix operand order repaired and checked by reading, not compiled, not run on hardware
//
#include <hip/hip_runtime.h>

typedef __attribute__((ext_vector_type(16))) __bf16 v16bf;
typedef __attribute__((ext_vector_type(8)))  float  v8f;

union BF16Frag {
  v16bf v;
  unsigned short u[16];
  uint4 q[2];
};

__device__ __forceinline__ unsigned short f2bf(float f) {
  union { float f; unsigned u; } x; x.f = f;
  unsigned r = x.u + 0x7FFFu + ((x.u >> 16) & 1u);  // round-to-nearest-even
  return (unsigned short)(r >> 16);
}

#define LOG2E 1.442695041f

// Branch-free activations: single v_exp_f32 + v_rcp_f32 each, no libm, no EXEC divergence.
__device__ __forceinline__ float sigm(float v) {
  return __builtin_amdgcn_rcpf(1.0f + __builtin_amdgcn_exp2f(-v * LOG2E));
}
__device__ __forceinline__ float tanh_fast(float v) {
  // tanh(x) = 1 - 2/(exp(2x)+1); exp2 saturates to 0/inf in HW, rcp(inf)=0 -> clean +-1 tails
  return 1.0f - 2.0f * __builtin_amdgcn_rcpf(1.0f + __builtin_amdgcn_exp2f(2.0f * LOG2E * v));
}

#define T_LEN 512
#define D_IN  32
#define H_DIM 128
#define A_DIM 10
#define NKS   5            // K-steps of 32 over K = D_IN + H_DIM = 160
#define AST   168          // A-buffer row stride in bf16 elems (16B aligned, bank-spread)
#define GST   516          // gates row stride in floats (padded)

// out layout (flat): q[1024*10] | h_n[1024*128] | c_n[1024*128]
#define OUT_H 10240
#define OUT_C 141312

__global__ __launch_bounds__(512, 1)
void lstm_dqn_fused(const float* __restrict__ x,
                    const float* __restrict__ W_ih,
                    const float* __restrict__ W_hh,
                    const float* __restrict__ b_ih,
                    const float* __restrict__ b_hh,
                    const float* __restrict__ fc_w,
                    const float* __restrict__ fc_b,
                    float* __restrict__ out) {
  __shared__ unsigned short Abuf[16 * AST];   // bf16 [x_t(0..31) | h(32..159)] per batch row
  __shared__ float          Gbuf[16 * GST];   // f32 gates (i,f,g,o packed along 512 cols)

  const int tid  = threadIdx.x;
  const int lane = tid & 31;
  const int wave = tid >> 5;          // 0..15 : owns gate cols [wave*32, wave*32+32)
  const int r16  = lane & 15;
  const int hi   = lane >> 4;         // 0 or 1
  const int bg   = blockIdx.x;        // batch tile (16 rows)

  // ---- Preload weight B-fragments into registers (bf16), reused for all 512 steps ----
  // B fragment (this code's contiguous K order): lane n=lane&15 is column N, elems j=0..15 are
  // K = k0 + hi*16 + j. B[k][n] = W[n][k] since gates = [x,h] @ W^T.
  BF16Frag bfrag[NKS][2];
#pragma unroll
  for (int ks = 0; ks < NKS; ++ks) {
#pragma unroll
    for (int nt = 0; nt < 2; ++nt) {
      const int n  = wave * 32 + nt * 16 + r16;     // gate row 0..511
      const int k0 = ks * 32 + hi * 16;             // start of 16 consecutive K
      const float* src = (k0 < D_IN) ? (W_ih + n * D_IN + k0)
                                     : (W_hh + n * H_DIM + (k0 - D_IN));
#pragma unroll
      for (int j = 0; j < 16; ++j)
        bfrag[ks][nt].u[j] = f2bf(src[j]);
    }
  }

  // ---- Per-thread elementwise ownership: 4 consecutive h-cols of one batch row ----
  const int brow  = tid >> 5;          // 0..15
  const int hbase = (tid & 31) * 4;    // 0..124
  float bias[4][4];                    // [gate][i] = b_ih + b_hh
#pragma unroll
  for (int g = 0; g < 4; ++g)
#pragma unroll
    for (int i = 0; i < 4; ++i) {
      const int idx = g * H_DIM + hbase + i;
      bias[g][i] = b_ih[idx] + b_hh[idx];
    }

  float c[4]    = {0.f, 0.f, 0.f, 0.f};
  float hreg[4] = {0.f, 0.f, 0.f, 0.f};

  // ---- Init A-buffer: h = 0, stage x_0 ----
#pragma unroll
  for (int i = 0; i < 4; ++i)
    Abuf[brow * AST + D_IN + hbase + i] = 0;
  {
    const int d = tid & 31;
    Abuf[brow * AST + d] = f2bf(x[(bg * 16 + brow) * (T_LEN * D_IN) + 0 * D_IN + d]);
  }
  __syncthreads();

  // ---- Recurrent scan ----
  for (int t = 0; t < T_LEN; ++t) {
    v8f acc0 = {};
    v8f acc1 = {};
#pragma unroll
    for (int ks = 0; ks < NKS; ++ks) {
      // A operand (16-bit, 16x32): lane row r16, elems j=0..15 -> K = k0 + hi*16 + j
      //                                      (same contiguous K order as B)
      BF16Frag afrag;
      const int base = r16 * AST + ks * 32 + hi * 16;  // 16B aligned
      afrag.q[0] = *(const uint4*)&Abuf[base];
      afrag.q[1] = *(const uint4*)&Abuf[base + 8];
      acc0 = __builtin_amdgcn_wmma_f32_16x16x32_bf16(
          false, afrag.v, false, bfrag[ks][0].v, (short)0, acc0, false, false);
      acc1 = __builtin_amdgcn_wmma_f32_16x16x32_bf16(
          false, afrag.v, false, bfrag[ks][1].v, (short)0, acc1, false, false);
    }

    // Scatter C/D fragment to LDS gates: elem e -> M = e + 8*hi, N = wave*32 + nt*16 + r16
    const int n0 = wave * 32 + r16;
#pragma unroll
    for (int e = 0; e < 8; ++e) {
      const int m = e + 8 * hi;
      Gbuf[m * GST + n0]      = acc0[e];
      Gbuf[m * GST + n0 + 16] = acc1[e];
    }
    __syncthreads();

    // Elementwise LSTM cell update (gate order i,f,g,o along 4H)
#pragma unroll
    for (int i = 0; i < 4; ++i) {
      const int h = hbase + i;
      const float iv = Gbuf[brow * GST + h]             + bias[0][i];
      const float fv = Gbuf[brow * GST + H_DIM + h]     + bias[1][i];
      const float gv = Gbuf[brow * GST + 2 * H_DIM + h] + bias[2][i];
      const float ov = Gbuf[brow * GST + 3 * H_DIM + h] + bias[3][i];
      const float ig = sigm(iv);
      const float fg = sigm(fv);
      const float gg = tanh_fast(gv);
      const float og = sigm(ov);
      c[i] = fg * c[i] + ig * gg;
      const float hn = og * tanh_fast(c[i]);
      hreg[i] = hn;
      Abuf[brow * AST + D_IN + h] = f2bf(hn);
    }
    // Stage x_{t+1} (clamped index: branchless, keeps a single loop body)
    {
      const int tn = (t + 1 < T_LEN) ? (t + 1) : t;
      const int d  = tid & 31;
      Abuf[brow * AST + d] =
          f2bf(x[(bg * 16 + brow) * (T_LEN * D_IN) + tn * D_IN + d]);
    }
    __syncthreads();
  }

  // ---- Epilogue: h_n, c_n to global; f32 h copy to LDS for the tiny FC ----
  const int gb = bg * 16 + brow;
#pragma unroll
  for (int i = 0; i < 4; ++i) {
    const int h = hbase + i;
    out[OUT_H + gb * H_DIM + h] = hreg[i];
    out[OUT_C + gb * H_DIM + h] = c[i];
    Gbuf[brow * GST + h] = hreg[i];
  }
  __syncthreads();

  // q = h_n @ fc_w^T + fc_b : 16x10 per workgroup
  if (tid < 16 * A_DIM) {
    const int b = tid / A_DIM;
    const int a = tid % A_DIM;
    float s = fc_b[a];
#pragma unroll 4
    for (int h = 0; h < H_DIM; ++h)
      s += Gbuf[b * GST + h] * fc_w[a * H_DIM + h];
    out[(bg * 16 + b) * A_DIM + a] = s;
  }
}

extern "C" void kernel_launch(void* const* d_in, const int* in_sizes, int n_in,
                              void* d_out, int out_size, void* d_ws, size_t ws_size,
                              hipStream_t stream) {
  const float* x    = (const float*)d_in[0];
  const float* W_ih = (const float*)d_in[1];
  const float* W_hh = (const float*)d_in[2];
  const float* b_ih = (const float*)d_in[3];
  const float* b_hh = (const float*)d_in[4];
  const float* fc_w = (const float*)d_in[5];
  const float* fc_b = (const float*)d_in[6];
  float* out = (float*)d_out;

  // B=1024 -> 64 workgroups of 16 batch rows; 512 threads = 16 wave32 per WG
  lstm_dqn_fused<<<64, 512, 0, stream>>>(x, W_ih, W_hh, b_ih, b_hh, fc_w, fc_b, out);
}
